// GcnDecoderGraph_18167711662120
// MI455X (gfx1250) — matrix operand order repaired and checked by reading, not compiled, not run on hardware
//
#include <hip/hip_runtime.h>
#include <hip/hip_bf16.h>

typedef __bf16 bf16;
typedef __attribute__((ext_vector_type(16))) __bf16 v16bf;
typedef __attribute__((ext_vector_type(8)))  float   v8f;

#define NN   2048
#define BB   16
#define KC   256     // K-chunk staged in LDS per buffer
#define APAD 8       // 16B row pad to rotate LDS banks

union FragU { v16bf v; uint4 q[2]; };

static __device__ inline v8f wmma_bf16(v16bf a, v16bf b, v8f c) {
  // D = A(16x32 bf16) x B(32x16 bf16) + C(16x16 f32)
  return __builtin_amdgcn_wmma_f32_16x16x32_bf16(
      /*neg_a=*/false, a, /*neg_b=*/false, b,
      /*c_mod=*/(short)0, c, /*reuse_a=*/false, /*reuse_b=*/false);
}

// 32-bit LDS byte offset of a shared-memory pointer (addrspace(3) cast).
static __device__ inline unsigned lds_off_of(const void* p) {
  return (unsigned)(unsigned long long)
      (__attribute__((address_space(3))) const void*)p;
}

// One instruction moves 16B per lane, Memory -> LDS, tracked by ASYNCcnt.
static __device__ inline void async_ld_16B(unsigned lds_off,
                                           const bf16* gptr) {
  asm volatile("global_load_async_to_lds_b128 %0, %1, off"
               :: "v"(lds_off), "v"((unsigned long long)gptr)
               : "memory");
}

static __device__ inline void wait_asynccnt0() {
  asm volatile("s_wait_asynccnt 0x0" ::: "memory");
}

// Issue async DMA of a 16 x KC bf16 tile (rows m0..m0+15, cols k0..k0+KC)
// from Ab (row-major, stride NN) into lds[16][KC+APAD].
static __device__ inline void async_copy_tile(const bf16* __restrict__ Ab,
                                              bf16 (*lds)[KC + APAD],
                                              int m0, int k0,
                                              int tid, int nthr) {
  for (int idx = tid; idx < 16 * (KC / 8); idx += nthr) {
    int row = idx / (KC / 8);
    int chk = idx % (KC / 8);
    async_ld_16B(lds_off_of(&lds[row][chk * 8]),
                 Ab + (size_t)(m0 + row) * NN + k0 + chk * 8);
  }
}

// ---------------------------------------------------------------------------
// S = A (B,N,N bf16 row-major, = adj^T) x X, with X given feature-major
// BT (B,C,N). Output also feature-major CT (B,C,N) in bf16 for the next GEMM.
// One workgroup: 16 node rows x all C features (one wave per 16 features).
// A stream is staged via double-buffered async-to-LDS DMA.
// ---------------------------------------------------------------------------
__global__ __launch_bounds__(128) void gemm_spmm_store(
    const bf16* __restrict__ A, const bf16* __restrict__ BT,
    bf16* __restrict__ CT, int C)
{
  const int nthr = blockDim.x;
  const int tid  = threadIdx.x;
  const int wave = tid >> 5;
  const int lane = tid & 31;
  const int half = lane >> 4;
  const int l15  = lane & 15;
  const int m0   = blockIdx.x * 16;
  const int b    = blockIdx.y;
  const int d0   = wave * 16;

  const bf16* Ab  = A  + (size_t)b * NN * NN;
  const bf16* BTb = BT + (size_t)b * C * NN;
  bf16*       CTb = CT + (size_t)b * C * NN;

  __shared__ __align__(16) bf16 Alds[2][16][KC + APAD];

  v8f acc = {};

  // prologue: DMA chunk 0 into buffer 0
  async_copy_tile(Ab, Alds[0], m0, 0, tid, nthr);
  wait_asynccnt0();
  __syncthreads();

  int cur = 0;
  for (int k0 = 0; k0 < NN; k0 += KC) {
    // overlap: DMA next chunk into the other buffer while we compute
    if (k0 + KC < NN)
      async_copy_tile(Ab, Alds[cur ^ 1], m0, k0 + KC, tid, nthr);

    for (int kk = 0; kk < KC; kk += 32) {
      FragU af, bfr;
      // A-frag: lane = row (l15); K chunks at half*8 and 16+half*8
      af.q[0] = *(const uint4*)&Alds[cur][l15][kk + half * 8];
      af.q[1] = *(const uint4*)&Alds[cur][l15][kk + 16 + half * 8];
      // B-frag: lane = column (feature d0+l15); K chunks at half*8 and 16+half*8
      const bf16* bp = BTb + (size_t)(d0 + l15) * NN + k0 + kk + half * 8;
      bfr.q[0] = *(const uint4*)bp;
      bfr.q[1] = *(const uint4*)(bp + 16);
      acc = wmma_bf16(af.v, bfr.v, acc);
    }
    wait_asynccnt0();
    __syncthreads();
    cur ^= 1;
  }

  // D layout: VGPR r -> row m0 + r + 8*half, col (feature) d0 + l15
  #pragma unroll
  for (int r = 0; r < 8; ++r)
    CTb[(size_t)(d0 + l15) * NN + m0 + r + 8 * half] = (bf16)acc[r];
}

// ---------------------------------------------------------------------------
// T = inv_deg x S (WMMA), then fused layer epilogue:
//   h = T + x ; out = h*W + x*Bm ; out /= max(||out||_2, eps) ; leaky_relu
// Writes next-layer fp32 node-major and bf16 feature-major copies.
// ---------------------------------------------------------------------------
__global__ __launch_bounds__(128) void gemm_combine(
    const bf16*  __restrict__ A,    // inv_deg bf16 (B,N,N)
    const bf16*  __restrict__ ST,   // S feature-major (B,Cin,N)
    const float* __restrict__ Xin,  // layer input fp32 (B,N,Cin)
    const float* __restrict__ W,    // (Cin,Cout)
    const float* __restrict__ Bm,   // (Cin,Cout)
    float* __restrict__ Xout,       // (B,N,Cout)
    bf16*  __restrict__ XTout,      // (B,Cout,N)
    int Cin, int Cout)
{
  const int nthr = blockDim.x;
  const int tid  = threadIdx.x;
  const int wave = tid >> 5;
  const int lane = tid & 31;
  const int half = lane >> 4;
  const int l15  = lane & 15;
  const int m0   = blockIdx.x * 16;
  const int b    = blockIdx.y;
  const int d0   = wave * 16;

  const bf16* Ab  = A  + (size_t)b * NN * NN;
  const bf16* BTb = ST + (size_t)b * Cin * NN;

  __shared__ __align__(16) bf16 Alds[2][16][KC + APAD];
  __shared__ float Hl[16][65];     // T, then h = T + x   (Cin <= 64)
  __shared__ float Xl[16][65];     // x tile
  __shared__ float Ol[16][129];    // out tile            (Cout <= 128)
  __shared__ float innorm[16];

  v8f acc = {};

  async_copy_tile(Ab, Alds[0], m0, 0, tid, nthr);
  wait_asynccnt0();
  __syncthreads();

  int cur = 0;
  for (int k0 = 0; k0 < NN; k0 += KC) {
    if (k0 + KC < NN)
      async_copy_tile(Ab, Alds[cur ^ 1], m0, k0 + KC, tid, nthr);

    for (int kk = 0; kk < KC; kk += 32) {
      FragU af, bfr;
      af.q[0] = *(const uint4*)&Alds[cur][l15][kk + half * 8];
      af.q[1] = *(const uint4*)&Alds[cur][l15][kk + 16 + half * 8];
      const bf16* bp = BTb + (size_t)(d0 + l15) * NN + k0 + kk + half * 8;
      bfr.q[0] = *(const uint4*)bp;
      bfr.q[1] = *(const uint4*)(bp + 16);
      acc = wmma_bf16(af.v, bfr.v, acc);
    }
    wait_asynccnt0();
    __syncthreads();
    cur ^= 1;
  }

  // spill accumulator tile to LDS (node-major)
  #pragma unroll
  for (int r = 0; r < 8; ++r)
    Hl[r + 8 * half][d0 + l15] = acc[r];
  __syncthreads();

  // h = T + x, keep x for the Bm term
  for (int idx = tid; idx < 16 * Cin; idx += nthr) {
    int i = idx / Cin, k = idx % Cin;
    float xv = Xin[((size_t)b * NN + (m0 + i)) * Cin + k];
    Xl[i][k] = xv;
    Hl[i][k] += xv;
  }
  __syncthreads();

  // out = h*W + x*Bm
  for (int idx = tid; idx < 16 * Cout; idx += nthr) {
    int i = idx / Cout, j = idx % Cout;
    float s = 0.f;
    for (int k = 0; k < Cin; ++k)
      s += Hl[i][k] * W[k * Cout + j] + Xl[i][k] * Bm[k * Cout + j];
    Ol[i][j] = s;
  }
  __syncthreads();

  if (tid < 16) {
    float ss = 0.f;
    for (int j = 0; j < Cout; ++j) ss += Ol[tid][j] * Ol[tid][j];
    innorm[tid] = 1.f / fmaxf(sqrtf(ss), 1e-12f);
  }
  __syncthreads();

  for (int idx = tid; idx < 16 * Cout; idx += nthr) {
    int i = idx / Cout, j = idx % Cout;
    float v = Ol[i][j] * innorm[i];
    v = (v >= 0.f) ? v : 0.1f * v;                 // leaky_relu(0.1)
    Xout[((size_t)b * NN + (m0 + i)) * Cout + j] = v;
    XTout[(size_t)b * Cout * NN + (size_t)j * NN + (m0 + i)] = (bf16)v;
  }
}

// ---------------------------------------------------------------------------
// Pre-pass conversions
// ---------------------------------------------------------------------------
__global__ void transpose_f32_to_bf16(const float* __restrict__ in,
                                      bf16* __restrict__ out, int N)
{
  __shared__ float tile[32][33];
  const int b  = blockIdx.z;
  const int x0 = blockIdx.x * 32, y0 = blockIdx.y * 32;
  const int tx = threadIdx.x, ty = threadIdx.y;
  const float* src = in + (size_t)b * N * N;
  bf16* dst        = out + (size_t)b * N * N;
  for (int j = 0; j < 32; j += 8)
    tile[ty + j][tx] = src[(size_t)(y0 + ty + j) * N + x0 + tx];
  __syncthreads();
  for (int j = 0; j < 32; j += 8)
    dst[(size_t)(x0 + ty + j) * N + y0 + tx] = (bf16)tile[tx][ty + j];
}

__global__ void convert_f32_to_bf16(const float* __restrict__ in,
                                    bf16* __restrict__ out, size_t n)
{
  size_t stride = (size_t)gridDim.x * blockDim.x;
  for (size_t i = (size_t)blockIdx.x * blockDim.x + threadIdx.x; i < n; i += stride)
    out[i] = (bf16)in[i];
}

__global__ void transpose_x_kernel(const float* __restrict__ x,
                                   bf16* __restrict__ xt, int N, int C)
{
  size_t total  = (size_t)BB * C * N;
  size_t stride = (size_t)gridDim.x * blockDim.x;
  for (size_t idx = (size_t)blockIdx.x * blockDim.x + threadIdx.x;
       idx < total; idx += stride) {
    size_t n = idx % N;
    size_t r = idx / N;
    size_t d = r % C;
    size_t b = r / C;
    xt[idx] = (bf16)x[(b * N + n) * C + d];
  }
}

// ---------------------------------------------------------------------------
// Decoder: a=h[:,N-2], b=h[:,N-1]; out = sum((a p1 p2) * (b p1), axis=1)
// One 128-thread block per batch (E = D = 128).
// ---------------------------------------------------------------------------
__global__ __launch_bounds__(128) void decoder_kernel(
    const float* __restrict__ X, const float* __restrict__ p1,
    const float* __restrict__ p2, float* __restrict__ out, int N, int E, int D)
{
  const int b = blockIdx.x;
  const int t = threadIdx.x;
  const float* av = X + ((size_t)b * N + (N - 2)) * E;
  const float* bv = X + ((size_t)b * N + (N - 1)) * E;
  __shared__ float t1[128], u[128], red[128];
  float s1 = 0.f, s2 = 0.f;
  for (int e = 0; e < E; ++e) {
    float pe = p1[e * D + t];
    s1 += av[e] * pe;
    s2 += bv[e] * pe;
  }
  t1[t] = s1; u[t] = s2;
  __syncthreads();
  float tv = 0.f;
  for (int k = 0; k < D; ++k) tv += t1[k] * p2[k * D + t];
  red[t] = tv * u[t];
  __syncthreads();
  for (int s = 64; s > 0; s >>= 1) {
    if (t < s) red[t] += red[t + s];
    __syncthreads();
  }
  if (t == 0) out[b] = red[0];
}

// ---------------------------------------------------------------------------
extern "C" void kernel_launch(void* const* d_in, const int* in_sizes, int n_in,
                              void* d_out, int out_size, void* d_ws, size_t ws_size,
                              hipStream_t stream) {
  (void)in_sizes; (void)n_in; (void)out_size; (void)ws_size;
  const float* x      = (const float*)d_in[0];
  const float* adj    = (const float*)d_in[1];
  const float* invdeg = (const float*)d_in[2];
  const float* w1 = (const float*)d_in[3];
  const float* b1 = (const float*)d_in[4];
  const float* w2 = (const float*)d_in[5];
  const float* b2 = (const float*)d_in[6];
  const float* w3 = (const float*)d_in[7];
  const float* b3 = (const float*)d_in[8];
  const float* p1 = (const float*)d_in[9];
  const float* p2 = (const float*)d_in[10];
  float* out = (float*)d_out;

  char* ws = (char*)d_ws;
  size_t off = 0;
  auto alloc = [&](size_t bytes) {
    void* p = ws + off;
    off = (off + bytes + 255) & ~(size_t)255;
    return p;
  };
  bf16*  adjT = (bf16*) alloc((size_t)BB * NN * NN * 2);   // 134 MB, adj^T bf16
  bf16*  idg  = (bf16*) alloc((size_t)BB * NN * NN * 2);   // 134 MB, inv_deg bf16
  bf16*  ST   = (bf16*) alloc((size_t)BB * 64  * NN * 2);  // S, feature-major
  bf16*  XT_A = (bf16*) alloc((size_t)BB * 128 * NN * 2);  // feature-major bf16 ping
  bf16*  XT_B = (bf16*) alloc((size_t)BB * 128 * NN * 2);  // feature-major bf16 pong
  float* XF_A = (float*)alloc((size_t)BB * NN * 128 * 4);  // node-major fp32 ping
  float* XF_B = (float*)alloc((size_t)BB * NN * 128 * 4);  // node-major fp32 pong

  // pre-pass: one-time bf16 conversions (halves all subsequent adj traffic)
  transpose_f32_to_bf16<<<dim3(NN/32, NN/32, BB), dim3(32, 8), 0, stream>>>(adj, adjT, NN);
  convert_f32_to_bf16<<<2048, 256, 0, stream>>>(invdeg, idg, (size_t)BB * NN * NN);
  transpose_x_kernel<<<512, 256, 0, stream>>>(x, XT_A, NN, 16);

  dim3 grid(NN / 16, BB);
  // layer 1 (Cin=16 -> 1 wave; Cout=64)
  gemm_spmm_store<<<grid, 32,  0, stream>>>(adjT, XT_A, ST, 16);
  gemm_combine  <<<grid, 32,  0, stream>>>(idg, ST, x,    w1, b1, XF_A, XT_B, 16, 64);
  // layer 2 (Cin=64 -> 4 waves; Cout=64)
  gemm_spmm_store<<<grid, 128, 0, stream>>>(adjT, XT_B, ST, 64);
  gemm_combine  <<<grid, 128, 0, stream>>>(idg, ST, XF_A, w2, b2, XF_B, XT_A, 64, 64);
  // layer 3 (Cin=64 -> 4 waves; Cout=128)
  gemm_spmm_store<<<grid, 128, 0, stream>>>(adjT, XT_A, ST, 64);
  gemm_combine  <<<grid, 128, 0, stream>>>(idg, ST, XF_B, w3, b3, XF_A, XT_B, 64, 128);
  // decoder
  decoder_kernel<<<BB, 128, 0, stream>>>(XF_A, p1, p2, out, NN, 128, 128);
}
